// ABRLLM_82643760710364
// MI455X (gfx1250) — matrix operand order repaired and checked by reading, not compiled, not run on hardware
//
#include <hip/hip_runtime.h>
#include <hip/hip_bf16.h>

// ---------------------------------------------------------------------------
// Decision-transformer fused block for MI455X (gfx1250, wave32, WMMA).
//   B=16 S=512 E=768 H=12 HD=64 HID=768 ; N = 8192 tokens, 8 slots/token.
// Pipeline:
//   K1: pack 7 weight matrices fp32 -> bf16, transposed (N-major) in d_ws.
//   K2: fully fused per-8-token tile: stacked->LDS, QKV WMMA, per-head
//       masked softmax attention, Wo WMMA + residual + LN, fusion,
//       Ws/Wa/Wr WMMA + leaky-relu (+LN for state), outputs to d_out.
// ---------------------------------------------------------------------------

typedef __attribute__((ext_vector_type(16))) __bf16     v16bf;
typedef __attribute__((ext_vector_type(8)))  float      v8f;
typedef __attribute__((ext_vector_type(4)))  unsigned int u32x4;

static constexpr int  EDIM   = 768;
static constexpr int  NHEAD  = 12;
static constexpr int  HDIM   = 64;
static constexpr int  TOK    = 8;           // tokens per workgroup
static constexpr int  MROWS  = 64;          // TOK * 8 slots
static constexpr int  KSTEPS = EDIM / 32;   // 24 wmma k-steps
static constexpr long NTOK   = 8192;        // B*S
static constexpr long OUTSZ  = NTOK * (long)EDIM;

// LDS pool layout (bytes)
static constexpr int XS_LD   = 776;                       // row stride (bf16 elems), pad 8
static constexpr int XS_OFF  = 0;
static constexpr int XS_BYT  = MROWS * XS_LD * 2;         // 99328
static constexpr int AO_OFF  = XS_OFF + XS_BYT;           // 99328
static constexpr int AO_BYT  = MROWS * XS_LD * 2;         // 99328 (aliased by Z later)
static constexpr int HB_LD   = 72;                        // 64 + 8 pad
static constexpr int HB_OFF  = AO_OFF + AO_BYT;           // 198656
static constexpr int HB_ONE  = MROWS * HB_LD * 2;         // 9216
static constexpr int HB_BYT  = 3 * HB_ONE;                // 27648 (aliased by Sbuf fp32 8x776)
static constexpr int SC_OFF  = HB_OFF + HB_BYT;           // 226304 : 8 waves * 64 f32
static constexpr int RED_OFF = SC_OFF + 2048;             // 228352 : 64*8 f32
static constexpr int MU_OFF  = RED_OFF + 2048;            // 230400 : 64 f32
static constexpr int RS_OFF  = MU_OFF + 256;              // 230656 : 64 f32
static constexpr int LDS_TOT = RS_OFF + 256;              // 230912 bytes (< 320KB WGP LDS)

union BFrag { v16bf v; u32x4 q[2]; };

// A fragment 16x32 bf16 from LDS (row-major, stride ld elements).
// lanes 0-15: M=lane, K 0-7 then 16-23 ; lanes 16-31: M=lane-16, K 8-15 then 24-31.
__device__ inline v16bf load_a_frag(const __bf16* base, int ld, int mbase, int kbase, int lane) {
    int row = mbase + (lane & 15);
    int k0  = kbase + ((lane >> 4) << 3);         // +0 or +8
    const __bf16* p = base + row * ld + k0;
    BFrag f;
    f.q[0] = *(const u32x4*)(p);
    f.q[1] = *(const u32x4*)(p + 16);
    return f.v;
}

// B fragment 32x16 bf16 from packed transposed weights WT[N][K] (K-major rows).
// lanes 0-15: N=lane, K 0-7 then 16-23 ; lanes 16-31: N=lane-16, K 8-15 then 24-31.
__device__ inline v16bf load_b_frag(const __bf16* WT, int ncol, int kbase, int lane) {
    int n  = ncol + (lane & 15);
    int k0 = kbase + ((lane >> 4) << 3);          // +0 or +8
    const __bf16* p = WT + (size_t)n * EDIM + k0;
    BFrag f;
    f.q[0] = *(const u32x4*)(p);
    f.q[1] = *(const u32x4*)(p + 16);
    return f.v;
}

__device__ inline v8f wmma_bf16(v16bf a, v16bf b, v8f c) {
    return __builtin_amdgcn_wmma_f32_16x16x32_bf16(false, a, false, b, (short)0, c, false, false);
}

// ---------------------------------------------------------------------------
// K1: fp32 W[K][N] -> bf16 WT[N][K] for q,k,v,o,s,a,r (contiguous in d_ws)
// ---------------------------------------------------------------------------
__global__ __launch_bounds__(256) void ABR_pack_weights(
    const float* __restrict__ Wq, const float* __restrict__ Wk,
    const float* __restrict__ Wv, const float* __restrict__ Wo,
    const float* __restrict__ Ws, const float* __restrict__ Wa,
    const float* __restrict__ Wr, __bf16* __restrict__ Wp) {
    const float* src[7] = {Wq, Wk, Wv, Wo, Ws, Wa, Wr};
    const int mat = blockIdx.y;
    const float* W = src[mat];
    __bf16* dst = Wp + (size_t)mat * EDIM * EDIM;
    for (int i = blockIdx.x * 256 + threadIdx.x; i < EDIM * EDIM; i += gridDim.x * 256) {
        int n = i / EDIM, k = i % EDIM;
        dst[i] = (__bf16)W[(size_t)k * EDIM + n];   // coalesced writes along k
    }
}

// ---------------------------------------------------------------------------
// K2: fused transformer block, one workgroup = 8 tokens (64x768 tile)
// ---------------------------------------------------------------------------
__global__ __launch_bounds__(256) void ABR_fused_block(
    const float* __restrict__ sf,   const float* __restrict__ act,
    const float* __restrict__ retE, const __bf16* __restrict__ Wp,
    const float* __restrict__ bq,   const float* __restrict__ bk,
    const float* __restrict__ bv,   const float* __restrict__ bo,
    const float* __restrict__ lng,  const float* __restrict__ lnb,
    const float* __restrict__ fw,
    const float* __restrict__ bs,   const float* __restrict__ ba,
    const float* __restrict__ br,
    const float* __restrict__ fng,  const float* __restrict__ fnb,
    float* __restrict__ out) {

    __shared__ __align__(16) unsigned char smem[LDS_TOT];
    __bf16* Xs  = (__bf16*)(smem + XS_OFF);     // stacked -> (out+res) -> enhanced
    __bf16* AO  = (__bf16*)(smem + AO_OFF);     // attention output (later: Z matrix)
    __bf16* Qh  = (__bf16*)(smem + HB_OFF);
    __bf16* Kh  = (__bf16*)(smem + HB_OFF + HB_ONE);
    __bf16* Vh  = (__bf16*)(smem + HB_OFF + 2 * HB_ONE);
    float*  SC  = (float*)(smem + SC_OFF);
    float*  RED = (float*)(smem + RED_OFF);
    float*  MU  = (float*)(smem + MU_OFF);
    float*  RS  = (float*)(smem + RS_OFF);
    float*  Sbuf = (float*)(smem + HB_OFF);     // alias: staged state rows (8 x 776 f32)

    const int tid  = threadIdx.x;
    const int wave = tid >> 5;
    const int lane = tid & 31;
    const long tok0 = (long)blockIdx.x * TOK;

    // ---- Step 1: gather stacked tile (64 x 768) fp32 -> bf16 LDS -----------
    for (int i = tid; i < MROWS * (EDIM / 4); i += 256) {
        int row = i / (EDIM / 4);
        int c   = (i % (EDIM / 4)) * 4;
        int t = row >> 3, s = row & 7;
        const float* p;
        if (s < 6)       p = sf   + ((tok0 + t) * 6 + s) * EDIM + c;
        else if (s == 6) p = act  + (tok0 + t) * EDIM + c;
        else             p = retE + (tok0 + t) * EDIM + c;
        float4 v = *(const float4*)p;
        __bf16* d = Xs + row * XS_LD + c;
        d[0] = (__bf16)v.x; d[1] = (__bf16)v.y; d[2] = (__bf16)v.z; d[3] = (__bf16)v.w;
    }
    __syncthreads();

    // ---- Step 2: per-head QKV (WMMA) + masked softmax attention ------------
    for (int h = 0; h < NHEAD; ++h) {
        // 48 tiles (3 mats x 4 mt x 4 nt), 6 per wave, serial acc
        for (int tt = 0; tt < 6; ++tt) {
            int id  = wave * 6 + tt;
            int mat = id >> 4;                 // 0=q 1=k 2=v
            int rem = id & 15;
            int mt = rem >> 2, nt = rem & 3;
            const __bf16* WT = Wp + (size_t)mat * EDIM * EDIM;
            int ncol = h * HDIM + nt * 16;
            v8f acc = {};
            for (int k = 0; k < KSTEPS; ++k) {
                v16bf a = load_a_frag(Xs, XS_LD, mt * 16, k * 32, lane);
                v16bf b = load_b_frag(WT, ncol, k * 32, lane);
                acc = wmma_bf16(a, b, acc);
            }
            const float* bias = (mat == 0) ? bq : ((mat == 1) ? bk : bv);
            __bf16* HB = (mat == 0) ? Qh : ((mat == 1) ? Kh : Vh);
            int n = lane & 15;
            float bval = bias[ncol + n];
            int rbase = mt * 16 + ((lane >> 4) ? 8 : 0);
            for (int r = 0; r < 8; ++r)
                HB[(rbase + r) * HB_LD + nt * 16 + n] = (__bf16)(acc[r] + bval);
        }
        __syncthreads();

        // attention: wave == token (8 waves, 8 tokens)
        {
            int t = wave;
            float* sc = SC + wave * 64;
            for (int e = lane; e < 64; e += 32) {
                int qs = e >> 3, ks = e & 7;
                const __bf16* qr = Qh + (t * 8 + qs) * HB_LD;
                const __bf16* kr = Kh + (t * 8 + ks) * HB_LD;
                float sum = 0.f;
                for (int d = 0; d < HDIM; ++d) sum += (float)qr[d] * (float)kr[d];
                sum *= 0.125f;                               // 1/sqrt(64)
                bool ok = (qs < 6) ? (ks < 6) : ((qs == 6) ? (ks < 7) : true);
                sc[e] = ok ? sum : -1e30f;
            }
            __syncthreads();
            if (lane < 8) {                                  // softmax row `lane`
                float mx = -1e30f;
                for (int j = 0; j < 8; ++j) mx = fmaxf(mx, sc[lane * 8 + j]);
                float ex[8], ssum = 0.f;
                for (int j = 0; j < 8; ++j) { ex[j] = __expf(sc[lane * 8 + j] - mx); ssum += ex[j]; }
                float inv = 1.f / ssum;
                for (int j = 0; j < 8; ++j) sc[lane * 8 + j] = ex[j] * inv;
            }
            __syncthreads();
            for (int e = lane; e < 512; e += 32) {           // attn @ V
                int qs = e >> 6, d = e & 63;
                float sum = 0.f;
                for (int ks = 0; ks < 8; ++ks)
                    sum += sc[qs * 8 + ks] * (float)Vh[(t * 8 + ks) * HB_LD + d];
                AO[(t * 8 + qs) * XS_LD + h * HDIM + d] = (__bf16)sum;
            }
        }
        __syncthreads();
    }

    // ---- Step 3: out = AO @ Wo + bo + residual, then LayerNorm -------------
    const __bf16* WTo = Wp + (size_t)3 * EDIM * EDIM;
    for (int nc = 0; nc < 4; ++nc) {                // N chunks of 192 (12 tiles)
        v8f acc[6] = {};
        for (int k = 0; k < KSTEPS; ++k) {
            for (int tt = 0; tt < 6; ++tt) {
                int id = wave * 6 + tt;
                int mt = id / 12, nt = id % 12;
                v16bf a = load_a_frag(AO, XS_LD, mt * 16, k * 32, lane);
                v16bf b = load_b_frag(WTo, nc * 192 + nt * 16, k * 32, lane);
                acc[tt] = wmma_bf16(a, b, acc[tt]);
            }
        }
        for (int tt = 0; tt < 6; ++tt) {
            int id = wave * 6 + tt;
            int mt = id / 12, nt = id % 12;
            int n  = nc * 192 + nt * 16 + (lane & 15);
            float bias = bo[n];
            int rbase = mt * 16 + ((lane >> 4) ? 8 : 0);
            for (int r = 0; r < 8; ++r) {
                int row = rbase + r;
                float x = acc[tt][r] + bias + (float)Xs[row * XS_LD + n]; // residual
                Xs[row * XS_LD + n] = (__bf16)x;   // unique owner: read-then-write
            }
        }
    }
    __syncthreads();

    // LayerNorm over 64 rows of Xs (768 elems each)
    {
        int row = tid >> 2, part = tid & 3;        // 4 partials per row
        float s = 0.f, s2 = 0.f;
        const __bf16* xr = Xs + row * XS_LD + part * 192;
        for (int c = 0; c < 192; ++c) { float v = (float)xr[c]; s += v; s2 += v * v; }
        RED[row * 8 + part] = s; RED[row * 8 + 4 + part] = s2;
    }
    __syncthreads();
    if (tid < 64) {
        float s  = RED[tid*8] + RED[tid*8+1] + RED[tid*8+2] + RED[tid*8+3];
        float s2 = RED[tid*8+4] + RED[tid*8+5] + RED[tid*8+6] + RED[tid*8+7];
        float mu = s * (1.f / 768.f);
        float var = s2 * (1.f / 768.f) - mu * mu;
        MU[tid] = mu; RS[tid] = rsqrtf(var + 1e-5f);
    }
    __syncthreads();
    for (int i = tid; i < MROWS * EDIM; i += 256) {
        int row = i / EDIM, c = i % EDIM;
        float v = ((float)Xs[row * XS_LD + c] - MU[row]) * RS[row] * lng[c] + lnb[c];
        Xs[row * XS_LD + c] = (__bf16)v;           // Xs := enhanced (bf16)
    }
    __syncthreads();

    // ---- Step 4: fusion + Ws/Wa/Wr projections -----------------------------
    float fwv[6];
    {
        float mx = -1e30f;
        for (int i = 0; i < 6; ++i) { fwv[i] = fw[i]; mx = fmaxf(mx, fwv[i]); }
        float s = 0.f;
        for (int i = 0; i < 6; ++i) { fwv[i] = __expf(fwv[i] - mx); s += fwv[i]; }
        float inv = 1.f / s;
        for (int i = 0; i < 6; ++i) fwv[i] *= inv;
    }
    __bf16* Z = AO;                                // alias: 48 x 768 (3 M-tiles, rows 8-15 zero)
    for (int i = tid; i < 48 * EDIM; i += 256) {
        int zr = i / EDIM, c = i % EDIM;
        int mt = zr >> 4, r = zr & 15;
        float v = 0.f;
        if (r < 8) {
            if (mt == 0) { for (int s = 0; s < 6; ++s) v += fwv[s] * (float)Xs[(r*8 + s) * XS_LD + c]; }
            else if (mt == 1) v = (float)Xs[(r*8 + 6) * XS_LD + c];
            else              v = (float)Xs[(r*8 + 7) * XS_LD + c];
        }
        Z[zr * XS_LD + c] = (__bf16)v;
    }
    __syncthreads();

    // 144 tiles (3 mt x 48 nt), 18 per wave, serial
    for (int tt = 0; tt < 18; ++tt) {
        int id = wave * 18 + tt;
        int mt = id / 48, nt = id % 48;
        const __bf16* WT = Wp + (size_t)(4 + mt) * EDIM * EDIM;   // Ws,Wa,Wr
        v8f acc = {};
        for (int k = 0; k < KSTEPS; ++k) {
            v16bf a = load_a_frag(Z, XS_LD, mt * 16, k * 32, lane);
            v16bf b = load_b_frag(WT, nt * 16, k * 32, lane);
            acc = wmma_bf16(a, b, acc);
        }
        const float* bias = (mt == 0) ? bs : ((mt == 1) ? ba : br);
        int n = nt * 16 + (lane & 15);
        float bval = bias[n];
        int rb = (lane >> 4) ? 8 : 0;
        for (int r = 0; r < 8; ++r) {
            int row = rb + r;                      // 0..15, valid rows < 8
            if (row < 8) {
                float y = acc[r] + bval;
                y = (y > 0.f) ? y : 0.01f * y;     // leaky_relu
                if (mt == 0) Sbuf[row * XS_LD + n] = y;    // state: stage for LN
                else out[(size_t)mt * OUTSZ + (tok0 + row) * EDIM + n] = y;
            }
        }
    }
    __syncthreads();

    // final LayerNorm on 8 staged state rows
    {
        int row = tid >> 5, part = tid & 31;       // 32 partials x 24 cols
        float s = 0.f, s2 = 0.f;
        const float* xr = Sbuf + row * XS_LD + part * 24;
        for (int c = 0; c < 24; ++c) { float v = xr[c]; s += v; s2 += v * v; }
        RED[row * 64 + part] = s; RED[row * 64 + 32 + part] = s2;
    }
    __syncthreads();
    if (tid < 8) {
        float s = 0.f, s2 = 0.f;
        for (int j = 0; j < 32; ++j) { s += RED[tid * 64 + j]; s2 += RED[tid * 64 + 32 + j]; }
        float mu = s * (1.f / 768.f);
        float var = s2 * (1.f / 768.f) - mu * mu;
        MU[tid] = mu; RS[tid] = rsqrtf(var + 1e-5f);
    }
    __syncthreads();
    for (int i = tid; i < 8 * EDIM; i += 256) {
        int row = i / EDIM, c = i % EDIM;
        float v = (Sbuf[row * XS_LD + c] - MU[row]) * RS[row] * fng[c] + fnb[c];
        out[(tok0 + row) * EDIM + c] = v;
    }
}

// ---------------------------------------------------------------------------
extern "C" void kernel_launch(void* const* d_in, const int* in_sizes, int n_in,
                              void* d_out, int out_size, void* d_ws, size_t ws_size,
                              hipStream_t stream) {
    (void)in_sizes; (void)n_in; (void)out_size; (void)ws_size;
    const float* sf   = (const float*)d_in[0];
    const float* act  = (const float*)d_in[1];
    const float* retE = (const float*)d_in[2];
    const float* Wq = (const float*)d_in[3];  const float* bq = (const float*)d_in[4];
    const float* Wk = (const float*)d_in[5];  const float* bk = (const float*)d_in[6];
    const float* Wv = (const float*)d_in[7];  const float* bv = (const float*)d_in[8];
    const float* Wo = (const float*)d_in[9];  const float* bo = (const float*)d_in[10];
    const float* lng = (const float*)d_in[11]; const float* lnb = (const float*)d_in[12];
    const float* fw  = (const float*)d_in[13];
    const float* Ws = (const float*)d_in[14]; const float* bs = (const float*)d_in[15];
    const float* Wa = (const float*)d_in[16]; const float* ba = (const float*)d_in[17];
    const float* Wr = (const float*)d_in[18]; const float* br = (const float*)d_in[19];
    const float* fng = (const float*)d_in[20]; const float* fnb = (const float*)d_in[21];
    float* out = (float*)d_out;
    __bf16* Wp = (__bf16*)d_ws;                 // 7 * 768*768 bf16 = 8.26 MB

    ABR_pack_weights<<<dim3((EDIM * EDIM) / 256, 7), 256, 0, stream>>>(
        Wq, Wk, Wv, Wo, Ws, Wa, Wr, Wp);

    ABR_fused_block<<<dim3(NTOK / TOK), 256, 0, stream>>>(
        sf, act, retE, Wp, bq, bk, bv, bo, lng, lnb, fw, bs, ba, br, fng, fnb, out);
}
